// Attn_block_76459007804152
// MI455X (gfx1250) — matrix operand order repaired and checked by reading, not compiled, not run on hardware
//
#include <hip/hip_runtime.h>

typedef __attribute__((ext_vector_type(16))) _Float16 v16h;
typedef __attribute__((ext_vector_type(8)))  _Float16 v8h;
typedef __attribute__((ext_vector_type(8)))  float    v8f;
typedef __attribute__((ext_vector_type(4)))  float    v4f;

#define BATCH 8
#define SEQ   4096
#define CH    64
#define NPB   (SEQ * CH)          // 262144 elements per batch
#define EPS   1e-5f
#define QBYTES ((size_t)BATCH * SEQ * CH * sizeof(_Float16))  // 4 MB
// softmax scale C^-0.5 folded with log2(e) so the kernel can use native exp2
#define QSCALE (0.125f * 1.44269504088896340736f)

static __device__ __forceinline__ float fast_exp2(float x) {
#if __has_builtin(__builtin_amdgcn_exp2f)
  return __builtin_amdgcn_exp2f(x);
#else
  return exp2f(x);
#endif
}

static __device__ __forceinline__ float fast_rcp(float x) {
#if __has_builtin(__builtin_amdgcn_rcpf)
  return __builtin_amdgcn_rcpf(x);
#else
  return 1.0f / x;
#endif
}

static __device__ __forceinline__ v8f wmma_f16(v16h a, v16h b, v8f c) {
  // D = A(16x32 f16) * B(32x16 f16) + C(16x16 f32)
  return __builtin_amdgcn_wmma_f32_16x16x32_f16(false, a, false, b, (short)0, c,
                                                false, false);
}

static __device__ __forceinline__ v8f vzero8() {
  v8f z;
#pragma unroll
  for (int i = 0; i < 8; ++i) z[i] = 0.0f;
  return z;
}

// A operand (16x32 f16): rows [0,16), cols [32t, 32t+32) of row-major f16 (ld).
// Layout: lanes 0-15 hold K 0..15, lanes 16-31 hold K 16..31 (matches B loaders).
static __device__ __forceinline__ v16h load_a16(const _Float16* base, int ld,
                                                int lane, int t) {
  const int row = lane & 15;
  const int kb  = 32 * t + 16 * (lane >> 4);
  const _Float16* p = base + row * ld + kb;
  v8h lo = *(const v8h*)(p);
  v8h hi = *(const v8h*)(p + 8);
  return __builtin_shufflevector(lo, hi, 0, 1, 2, 3, 4, 5, 6, 7,
                                 8, 9, 10, 11, 12, 13, 14, 15);
}

// B operand (K=32 x N=16) with B[k][n] = M[n][k], M row-major f16 (ld).
// Contiguous-k layout used here (same as load_a16): lane n (0-15) = col n, K 0..15; lane n+16 = col n, K 16..31.
static __device__ __forceinline__ v16h load_b16(const _Float16* base, int ld,
                                                int lane, int t) {
  const int col = lane & 15;
  const int kb  = 32 * t + 16 * (lane >> 4);
  return *(const v16h*)(base + col * ld + kb);   // 16 contiguous halves (32B)
}

// Same B operand but sourced from row-major f32 M, converted to f16.
static __device__ __forceinline__ v16h load_b32cvt(const float* base, int ld,
                                                   int lane, int t) {
  const int col = lane & 15;
  const int kb  = 32 * t + 16 * (lane >> 4);
  const float* p = base + col * ld + kb;
  v16h r;
#pragma unroll
  for (int i = 0; i < 16; ++i) r[i] = (_Float16)p[i];
  return r;
}

// ---------------------------------------------------------------------------
// Kernel 1: per-batch mean / rstd for GroupNorm(num_groups=1)
// ---------------------------------------------------------------------------
__global__ __launch_bounds__(256) void gn_stats(const float* __restrict__ x,
                                                float* __restrict__ stats) {
  __shared__ float s_sum[256];
  __shared__ float s_sq[256];
  const int b   = blockIdx.x;
  const int tid = threadIdx.x;
  const v4f* xb = (const v4f*)(x + (size_t)b * NPB);
  float s = 0.0f, ss = 0.0f;
  for (int i = tid; i < NPB / 4; i += 256) {
    v4f v = xb[i];
#pragma unroll
    for (int j = 0; j < 4; ++j) { s += v[j]; ss += v[j] * v[j]; }
  }
  s_sum[tid] = s;
  s_sq[tid]  = ss;
  __syncthreads();
  for (int off = 128; off > 0; off >>= 1) {
    if (tid < off) {
      s_sum[tid] += s_sum[tid + off];
      s_sq[tid]  += s_sq[tid + off];
    }
    __syncthreads();
  }
  if (tid == 0) {
    float mean = s_sum[0] * (1.0f / NPB);
    float var  = s_sq[0] * (1.0f / NPB) - mean * mean;
    stats[2 * b]     = mean;
    stats[2 * b + 1] = rsqrtf(var + EPS);
  }
}

// ---------------------------------------------------------------------------
// Kernel 2: h = (x - mu)*rs; q = (h Wq^T + bq) * QSCALE; k = h Wk^T + bk;
//           v = h Wv^T + bv.
// q,k stored f16 [B][S][C]; v stored TRANSPOSED f16 [B][C][S].
// One wave per 16 rows, 8 waves / block -> 128 rows per block.
// ---------------------------------------------------------------------------
__global__ __launch_bounds__(256) void qkv_proj(
    const float* __restrict__ x, const float* __restrict__ stats,
    const float* __restrict__ Wq, const float* __restrict__ bq,
    const float* __restrict__ Wk, const float* __restrict__ bk,
    const float* __restrict__ Wv, const float* __restrict__ bv,
    _Float16* __restrict__ qh, _Float16* __restrict__ kh,
    _Float16* __restrict__ vth) {
  const int wg   = blockIdx.x;        // BATCH * 32 blocks
  const int b    = wg >> 5;
  const int rblk = wg & 31;
  const int wave = threadIdx.x >> 5;
  const int lane = threadIdx.x & 31;
  const int row0 = rblk * 128 + wave * 16;

  const float mu = stats[2 * b];
  const float rs = stats[2 * b + 1];
  const float* xb = x + (size_t)b * NPB + (size_t)row0 * CH;

  // A operands: normalized x rows, f16
  v16h A0, A1;
  {
    const int row = lane & 15;
    const float* p = xb + row * CH;
#pragma unroll
    for (int t = 0; t < 2; ++t) {
      const int kb = 32 * t + 16 * (lane >> 4);
      v16h a;
#pragma unroll
      for (int i = 0; i < 8; ++i) a[i]     = (_Float16)((p[kb + i]      - mu) * rs);
#pragma unroll
      for (int i = 0; i < 8; ++i) a[8 + i] = (_Float16)((p[kb + 8 + i] - mu) * rs);
      if (t == 0) A0 = a; else A1 = a;
    }
  }

  const float* Ws[3] = {Wq, Wk, Wv};
  const float* bs[3] = {bq, bk, bv};
  const int colg   = lane & 15;
  const int rowadd = 8 * (lane >> 4);

#pragma unroll
  for (int mat = 0; mat < 3; ++mat) {
#pragma unroll
    for (int nt = 0; nt < 4; ++nt) {
      const float bias = bs[mat][nt * 16 + colg];
      v8f acc;
#pragma unroll
      for (int r = 0; r < 8; ++r) acc[r] = bias;
      v16h B0 = load_b32cvt(Ws[mat] + nt * 16 * CH, CH, lane, 0);
      v16h B1 = load_b32cvt(Ws[mat] + nt * 16 * CH, CH, lane, 1);
      acc = wmma_f16(A0, B0, acc);
      acc = wmma_f16(A1, B1, acc);
      if (mat == 0) {
        // fold softmax scale (incl. log2 e) into Q at f32 precision
        _Float16* dst = qh + ((size_t)b * SEQ + row0) * CH;
#pragma unroll
        for (int r = 0; r < 8; ++r)
          dst[(r + rowadd) * CH + nt * 16 + colg] = (_Float16)(acc[r] * QSCALE);
      } else if (mat == 1) {
        _Float16* dst = kh + ((size_t)b * SEQ + row0) * CH;
#pragma unroll
        for (int r = 0; r < 8; ++r)
          dst[(r + rowadd) * CH + nt * 16 + colg] = (_Float16)acc[r];
      } else {
        _Float16* dst = vth + (size_t)b * CH * SEQ;
#pragma unroll
        for (int r = 0; r < 8; ++r)
          dst[(size_t)(nt * 16 + colg) * SEQ + row0 + r + rowadd] = (_Float16)acc[r];
      }
    }
  }
}

// ---------------------------------------------------------------------------
// Kernel 3: flash attention (base-2 online softmax) + output projection +
// residual. One wave owns 16 query rows; keys processed in blocks of 64.
// ---------------------------------------------------------------------------
__global__ __launch_bounds__(256) void attn_kernel(
    const _Float16* __restrict__ qh, const _Float16* __restrict__ kh,
    const _Float16* __restrict__ vth, const float* __restrict__ x,
    const float* __restrict__ Wo, const float* __restrict__ bo,
    float* __restrict__ out) {
  __shared__ _Float16 smem[8 * 16 * 64];      // per-wave 16x64 f16 staging tile
  const int wg   = blockIdx.x;                // BATCH * 32 blocks
  const int b    = wg >> 5;
  const int rblk = wg & 31;
  const int wave = threadIdx.x >> 5;
  const int lane = threadIdx.x & 31;
  const int row0 = rblk * 128 + wave * 16;
  const int colg   = lane & 15;
  const int rowadd = 8 * (lane >> 4);

  const _Float16* qb = qh  + ((size_t)b * SEQ + row0) * CH;
  const _Float16* kb = kh  + (size_t)b * SEQ * CH;
  const _Float16* vb = vth + (size_t)b * CH * SEQ;
  _Float16* tile = smem + wave * (16 * 64);

  // Q already carries the softmax scale (folded in qkv_proj at f32 precision)
  const v16h Q0 = load_a16(qb, CH, lane, 0);
  const v16h Q1 = load_a16(qb, CH, lane, 1);

  v16h ones;
#pragma unroll
  for (int i = 0; i < 16; ++i) ones[i] = (_Float16)1.0f;

  float m[8];
#pragma unroll
  for (int r = 0; r < 8; ++r) m[r] = -1e30f;
  v8f lacc = vzero8();                        // running row sums (replicated)
  v8f O[4];
#pragma unroll
  for (int nt = 0; nt < 4; ++nt) O[nt] = vzero8();

  // per-lane prefetch addresses: one lane's K row == one 128B cacheline
  const _Float16* kpre = kb + (size_t)colg * CH;
  const _Float16* vpre = vb + (size_t)colg * SEQ + 16 * (lane >> 4);

  for (int j0 = 0; j0 < SEQ; j0 += 64) {
    // prefetch next key block (speculative; OOB addresses are dropped)
    __builtin_prefetch(kpre + (size_t)(j0 + 64) * CH, 0, 0);
    __builtin_prefetch(vpre + (j0 + 64), 0, 0);

    // --- scores S[16 x 64] = Q * K^T (f32 accum), 4 tiles of 16 keys -----
    v8f s[4];
#pragma unroll
    for (int jt = 0; jt < 4; ++jt) {
      const _Float16* kblk = kb + (size_t)(j0 + 16 * jt) * CH;
      v16h Bk0 = load_b16(kblk, CH, lane, 0);
      v16h Bk1 = load_b16(kblk, CH, lane, 1);
      s[jt] = wmma_f16(Q0, Bk0, vzero8());
      s[jt] = wmma_f16(Q1, Bk1, s[jt]);
    }

    // --- online softmax (base 2): row max over 16 lanes of each half-wave
    float tm[8];
#pragma unroll
    for (int r = 0; r < 8; ++r)
      tm[r] = fmaxf(fmaxf(s[0][r], s[1][r]), fmaxf(s[2][r], s[3][r]));
#pragma unroll
    for (int off = 1; off < 16; off <<= 1) {
#pragma unroll
      for (int r = 0; r < 8; ++r)
        tm[r] = fmaxf(tm[r], __shfl_xor(tm[r], off, 32));
    }

    float corr[8];
#pragma unroll
    for (int r = 0; r < 8; ++r) {
      const float mnew = fmaxf(m[r], tm[r]);
      corr[r] = fast_exp2(m[r] - mnew);
      m[r] = mnew;
    }
#pragma unroll
    for (int r = 0; r < 8; ++r) lacc[r] *= corr[r];
#pragma unroll
    for (int nt = 0; nt < 4; ++nt)
#pragma unroll
      for (int r = 0; r < 8; ++r) O[nt][r] *= corr[r];

    // --- P = exp2(S - m) -> f16 via LDS into A-operand layout ------------
#pragma unroll
    for (int jt = 0; jt < 4; ++jt)
#pragma unroll
      for (int r = 0; r < 8; ++r)
        tile[(r + rowadd) * 64 + 16 * jt + colg] =
            (_Float16)fast_exp2(s[jt][r] - m[r]);
    __builtin_amdgcn_wave_barrier();
    v16h P0 = load_a16(tile, 64, lane, 0);    // keys j0   .. j0+31
    v16h P1 = load_a16(tile, 64, lane, 1);    // keys j0+32 .. j0+63
    __builtin_amdgcn_wave_barrier();

    // row sums via WMMA against all-ones B (result replicated per column)
    lacc = wmma_f16(P0, ones, lacc);
    lacc = wmma_f16(P1, ones, lacc);

    // --- O += P * V  (V stored transposed: contiguous B operands) --------
#pragma unroll
    for (int nt = 0; nt < 4; ++nt) {
      const _Float16* vblk = vb + (size_t)(nt * 16) * SEQ + j0;
      v16h Bv0 = load_b16(vblk, SEQ, lane, 0);   // keys j0    .. j0+31
      v16h Bv1 = load_b16(vblk, SEQ, lane, 1);   // keys j0+32 .. j0+63
      O[nt] = wmma_f16(P0, Bv0, O[nt]);
      O[nt] = wmma_f16(P1, Bv1, O[nt]);
    }
  }

  // --- finalize: O /= l, stage to LDS as f16 A operand ---------------------
  float linv[8];
#pragma unroll
  for (int r = 0; r < 8; ++r) linv[r] = fast_rcp(lacc[r]);
#pragma unroll
  for (int nt = 0; nt < 4; ++nt)
#pragma unroll
    for (int r = 0; r < 8; ++r)
      tile[(r + rowadd) * 64 + nt * 16 + colg] = (_Float16)(O[nt][r] * linv[r]);
  __builtin_amdgcn_wave_barrier();
  v16h OA0 = load_a16(tile, 64, lane, 0);
  v16h OA1 = load_a16(tile, 64, lane, 1);

  // --- output projection + bias + residual ---------------------------------
  const float* xrow = x   + ((size_t)b * SEQ + row0) * CH;
  float*       orow = out + ((size_t)b * SEQ + row0) * CH;
#pragma unroll
  for (int nt = 0; nt < 4; ++nt) {
    const float bias = bo[nt * 16 + colg];
    v8f acc;
#pragma unroll
    for (int r = 0; r < 8; ++r) acc[r] = bias;
    v16h B0 = load_b32cvt(Wo + nt * 16 * CH, CH, lane, 0);
    v16h B1 = load_b32cvt(Wo + nt * 16 * CH, CH, lane, 1);
    acc = wmma_f16(OA0, B0, acc);
    acc = wmma_f16(OA1, B1, acc);
#pragma unroll
    for (int r = 0; r < 8; ++r) {
      const int rr = r + rowadd;
      const int cc = nt * 16 + colg;
      orow[rr * CH + cc] = xrow[rr * CH + cc] + acc[r];
    }
  }
}

// ---------------------------------------------------------------------------
extern "C" void kernel_launch(void* const* d_in, const int* in_sizes, int n_in,
                              void* d_out, int out_size, void* d_ws, size_t ws_size,
                              hipStream_t stream) {
  const float* x  = (const float*)d_in[0];
  // d_in[1] = temb (unused by reference math)
  const float* Wq = (const float*)d_in[2];
  const float* bq = (const float*)d_in[3];
  const float* Wk = (const float*)d_in[4];
  const float* bk = (const float*)d_in[5];
  const float* Wv = (const float*)d_in[6];
  const float* bv = (const float*)d_in[7];
  const float* Wo = (const float*)d_in[8];
  const float* bo = (const float*)d_in[9];
  float* out = (float*)d_out;

  char* ws = (char*)d_ws;
  _Float16* qh  = (_Float16*)(ws);
  _Float16* kh  = (_Float16*)(ws + QBYTES);
  _Float16* vth = (_Float16*)(ws + 2 * QBYTES);
  float*    stats = (float*)(ws + 3 * QBYTES);

  gn_stats<<<BATCH, 256, 0, stream>>>(x, stats);
  qkv_proj<<<BATCH * (SEQ / 128), 256, 0, stream>>>(x, stats, Wq, bq, Wk, bk,
                                                    Wv, bv, qh, kh, vth);
  attn_kernel<<<BATCH * (SEQ / 128), 256, 0, stream>>>(qh, kh, vth, x, Wo, bo, out);
}
